// MultiHeadSelfAttention_17420387352791
// MI455X (gfx1250) — matrix operand order repaired and checked by reading, not compiled, not run on hardware
//
#include <hip/hip_runtime.h>
#include <hip/hip_bf16.h>
#include <stdint.h>

#define BATCH  2
#define SEQ    2048
#define EMBED  1024
#define NHEAD  16
#define HDIM   64
#define MTOK   (BATCH * SEQ)   // 4096

typedef __bf16 v16bf __attribute__((ext_vector_type(16)));
typedef float  v8f   __attribute__((ext_vector_type(8)));
typedef unsigned int v4u __attribute__((ext_vector_type(4)));
typedef float  v4f   __attribute__((ext_vector_type(4)));

union FragBF {
    v16bf bf;
    v4u   q[2];
    unsigned short u[16];
};

__device__ __forceinline__ unsigned short f2bf(float f) {
    unsigned int u = __float_as_uint(f);
    u += 0x7FFFu + ((u >> 16) & 1u);          // round-to-nearest-even
    return (unsigned short)(u >> 16);
}

// ---------------------------------------------------------------------------
// 1. fp32 -> bf16 elementwise convert
// ---------------------------------------------------------------------------
__global__ void cvt_bf16_kernel(const float* __restrict__ in,
                                unsigned short* __restrict__ out, int n) {
    int i = blockIdx.x * blockDim.x + threadIdx.x;
    if (i < n) out[i] = f2bf(in[i]);
}

// ---------------------------------------------------------------------------
// 2. transpose-convert W[K,N] fp32 -> Wt[N,K] bf16 (LDS tiled, 32x32)
// ---------------------------------------------------------------------------
__global__ void transpose_cvt_kernel(const float* __restrict__ W,
                                     unsigned short* __restrict__ Wt) {
    __shared__ unsigned short tile[32][33];
    int x  = blockIdx.x * 32 + threadIdx.x;   // source column
    int y0 = blockIdx.y * 32;                 // source row base
    for (int j = threadIdx.y; j < 32; j += 8)
        tile[j][threadIdx.x] = f2bf(W[(size_t)(y0 + j) * EMBED + x]);
    __syncthreads();
    int xt  = blockIdx.y * 32 + threadIdx.x;  // dest column (= source row)
    int yt0 = blockIdx.x * 32;                // dest row base (= source col)
    for (int j = threadIdx.y; j < 32; j += 8)
        Wt[(size_t)(yt0 + j) * EMBED + xt] = tile[threadIdx.x][j];
}

// ---------------------------------------------------------------------------
// 3. bf16 WMMA GEMM: C[M,N] = A[M,K] * Bt[N,K]^T + bias
//    Each wave computes a 32x64 region: 2x4 grid of 16x16 WMMA tiles
//    (8 independent accumulators -> hides the 5-cycle WMMA->WMMA hazard,
//     A frags reused 4x, B frags reused 2x).
//    Block = 8 waves tiled 4(M) x 2(N) -> 128x128 per block.
//    MODE 0: bf16 store, MODE 1: bf16 transposed store (ldc = M),
//    MODE 2: fp32 store
//    Fragment layouts per CDNA5 ISA 7.12.2 (wave32):
//      A lane L (m=L&15, h=L>>4): K chunks [kk+8h .. +7], [kk+16+8h .. +7]
//      B lane L (n=L&15, h=L>>4): K chunks [kk+8h .. +7], [kk+16+8h .. +7] (from Bt row n)
// ---------------------------------------------------------------------------
template <int MODE>
__global__ void gemm_wmma_kernel(const unsigned short* __restrict__ A,
                                 const unsigned short* __restrict__ Bt,
                                 const float* __restrict__ bias,
                                 void* __restrict__ Cout,
                                 int Kdim, int ldc) {
    const int lane = threadIdx.x & 31;
    const int wave = threadIdx.x >> 5;
    const int mn   = lane & 15;
    const int half = lane >> 4;
    const int tileM = blockIdx.y * 128 + (wave >> 1) * 32;  // 2 M-tiles of 16
    const int tileN = blockIdx.x * 128 + (wave & 1) * 64;   // 4 N-tiles of 16

    const unsigned short* arow0 = A  + (size_t)(tileM + mn) * Kdim;
    const unsigned short* brow0 = Bt + (size_t)(tileN + mn) * Kdim;
    const size_t astride16 = (size_t)16 * Kdim;

    v8f acc[2][4];
#pragma unroll
    for (int i = 0; i < 2; ++i)
#pragma unroll
        for (int j = 0; j < 4; ++j)
            acc[i][j] = (v8f){0.f, 0.f, 0.f, 0.f, 0.f, 0.f, 0.f, 0.f};

    for (int kk = 0; kk < Kdim; kk += 32) {
        FragBF fa[2], fb[4];
#pragma unroll
        for (int i = 0; i < 2; ++i) {
            const unsigned short* ap = arow0 + i * astride16 + kk;
            fa[i].q[0] = *(const v4u*)(ap + 8 * half);
            fa[i].q[1] = *(const v4u*)(ap + 16 + 8 * half);
        }
#pragma unroll
        for (int j = 0; j < 4; ++j) {
            const unsigned short* bp = brow0 + j * astride16 + kk;
            fb[j].q[0] = *(const v4u*)(bp + 8 * half);
            fb[j].q[1] = *(const v4u*)(bp + 16 + 8 * half);
        }
#pragma unroll
        for (int i = 0; i < 2; ++i)
#pragma unroll
            for (int j = 0; j < 4; ++j)
                acc[i][j] = __builtin_amdgcn_wmma_f32_16x16x32_bf16(
                    false, fa[i].bf, false, fb[j].bf, (short)0, acc[i][j],
                    false, false);
    }

#pragma unroll
    for (int j = 0; j < 4; ++j) {
        const int col  = tileN + j * 16 + mn;
        const float bc = bias ? bias[col] : 0.f;
#pragma unroll
        for (int i = 0; i < 2; ++i) {
#pragma unroll
            for (int v = 0; v < 8; ++v) {
                const int row = tileM + i * 16 + v + 8 * half;
                const float val = acc[i][j][v] + bc;
                if (MODE == 0)
                    ((unsigned short*)Cout)[(size_t)row * ldc + col] = f2bf(val);
                else if (MODE == 1)
                    ((unsigned short*)Cout)[(size_t)col * ldc + row] = f2bf(val);
                else
                    ((float*)Cout)[(size_t)row * ldc + col] = val;
            }
        }
    }
}

// ---------------------------------------------------------------------------
// 4. scores = (Q_h K_h^T) / sqrt(HDIM), fp32 into d_out weights region.
//    Same 32x64-per-wave / 128x128-per-block tiling; K loop = HDIM (2 steps).
// ---------------------------------------------------------------------------
__global__ void scores_wmma_kernel(const unsigned short* __restrict__ Qb,
                                   const unsigned short* __restrict__ Kb,
                                   float* __restrict__ Wout) {
    const int lane = threadIdx.x & 31;
    const int wave = threadIdx.x >> 5;
    const int mn   = lane & 15;
    const int half = lane >> 4;
    const int bh = blockIdx.z;
    const int b = bh >> 4, h = bh & 15;
    const int tileM = blockIdx.y * 128 + (wave >> 1) * 32;
    const int tileN = blockIdx.x * 128 + (wave & 1) * 64;

    const unsigned short* arow0 =
        Qb + (size_t)(b * SEQ + tileM + mn) * EMBED + h * HDIM;
    const unsigned short* brow0 =
        Kb + (size_t)(b * SEQ + tileN + mn) * EMBED + h * HDIM;
    const size_t stride16 = (size_t)16 * EMBED;

    v8f acc[2][4];
#pragma unroll
    for (int i = 0; i < 2; ++i)
#pragma unroll
        for (int j = 0; j < 4; ++j)
            acc[i][j] = (v8f){0.f, 0.f, 0.f, 0.f, 0.f, 0.f, 0.f, 0.f};

#pragma unroll
    for (int kk = 0; kk < HDIM; kk += 32) {
        FragBF fa[2], fb[4];
#pragma unroll
        for (int i = 0; i < 2; ++i) {
            const unsigned short* ap = arow0 + i * stride16 + kk;
            fa[i].q[0] = *(const v4u*)(ap + 8 * half);
            fa[i].q[1] = *(const v4u*)(ap + 16 + 8 * half);
        }
#pragma unroll
        for (int j = 0; j < 4; ++j) {
            const unsigned short* bp = brow0 + j * stride16 + kk;
            fb[j].q[0] = *(const v4u*)(bp + 8 * half);
            fb[j].q[1] = *(const v4u*)(bp + 16 + 8 * half);
        }
#pragma unroll
        for (int i = 0; i < 2; ++i)
#pragma unroll
            for (int j = 0; j < 4; ++j)
                acc[i][j] = __builtin_amdgcn_wmma_f32_16x16x32_bf16(
                    false, fa[i].bf, false, fb[j].bf, (short)0, acc[i][j],
                    false, false);
    }

    float* obase = Wout + (size_t)bh * SEQ * SEQ;
#pragma unroll
    for (int i = 0; i < 2; ++i)
#pragma unroll
        for (int j = 0; j < 4; ++j)
#pragma unroll
            for (int v = 0; v < 8; ++v) {
                const int row = tileM + i * 16 + v + 8 * half;
                const int col = tileN + j * 16 + mn;
                obase[(size_t)row * SEQ + col] = acc[i][j][v] * 0.125f;
            }
}

// ---------------------------------------------------------------------------
// 5. row softmax in place, one block per row of [B*H*S, SEQ]; float4 accesses
// ---------------------------------------------------------------------------
__global__ void softmax_kernel(float* __restrict__ Wt) {
    __shared__ float red[256];
    v4f* p = (v4f*)(Wt + (size_t)blockIdx.x * SEQ);   // SEQ/4 = 512 vectors
    const int tid = threadIdx.x;

    float mx = -3.4e38f;
    for (int i = tid; i < SEQ / 4; i += 256) {
        v4f x = p[i];
        mx = fmaxf(mx, fmaxf(fmaxf(x[0], x[1]), fmaxf(x[2], x[3])));
    }
    red[tid] = mx;
    __syncthreads();
    for (int s = 128; s > 0; s >>= 1) {
        if (tid < s) red[tid] = fmaxf(red[tid], red[tid + s]);
        __syncthreads();
    }
    mx = red[0];
    __syncthreads();

    float sum = 0.f;
    for (int i = tid; i < SEQ / 4; i += 256) {
        v4f x = p[i];
        x[0] = __expf(x[0] - mx);
        x[1] = __expf(x[1] - mx);
        x[2] = __expf(x[2] - mx);
        x[3] = __expf(x[3] - mx);
        p[i] = x;
        sum += x[0] + x[1] + x[2] + x[3];
    }
    red[tid] = sum;
    __syncthreads();
    for (int s = 128; s > 0; s >>= 1) {
        if (tid < s) red[tid] += red[tid + s];
        __syncthreads();
    }
    const float inv = 1.0f / red[0];
    for (int i = tid; i < SEQ / 4; i += 256) {
        v4f x = p[i];
        x[0] *= inv; x[1] *= inv; x[2] *= inv; x[3] *= inv;
        p[i] = x;
    }
}

// ---------------------------------------------------------------------------
// 6. attn = P @ V_h. Wave covers 32 rows x all 64 head dims (2x4 acc tiles),
//    so the fp32->bf16 conversion of each A fragment is reused 4x.
//    Block = 8 waves stacked in M -> 256 rows. Output bf16 concat [MTOK,EMBED].
// ---------------------------------------------------------------------------
__global__ void av_wmma_kernel(const float* __restrict__ P,
                               const unsigned short* __restrict__ Vt,
                               unsigned short* __restrict__ concat) {
    const int lane = threadIdx.x & 31;
    const int wave = threadIdx.x >> 5;
    const int mn   = lane & 15;
    const int half = lane >> 4;
    const int bh = blockIdx.z;
    const int b = bh >> 4, h = bh & 15;
    const int tileM = blockIdx.y * 256 + wave * 32;

    const float* prow0 = P + ((size_t)bh * SEQ + tileM + mn) * SEQ;
    const unsigned short* brow0 =
        Vt + (size_t)(h * HDIM + mn) * MTOK + b * SEQ;
    const size_t bstride16 = (size_t)16 * MTOK;

    v8f acc[2][4];
#pragma unroll
    for (int i = 0; i < 2; ++i)
#pragma unroll
        for (int j = 0; j < 4; ++j)
            acc[i][j] = (v8f){0.f, 0.f, 0.f, 0.f, 0.f, 0.f, 0.f, 0.f};

    for (int kk = 0; kk < SEQ; kk += 32) {
        FragBF fa[2], fb[4];
#pragma unroll
        for (int i = 0; i < 2; ++i) {
            const float* pr = prow0 + (size_t)i * 16 * SEQ + kk;
            v4f x[4];
            x[0] = ((const v4f*)(pr + 8 * half))[0];
            x[1] = ((const v4f*)(pr + 8 * half))[1];
            x[2] = ((const v4f*)(pr + 16 + 8 * half))[0];
            x[3] = ((const v4f*)(pr + 16 + 8 * half))[1];
#pragma unroll
            for (int t = 0; t < 4; ++t)
#pragma unroll
                for (int e = 0; e < 4; ++e)
                    fa[i].u[4 * t + e] = f2bf(x[t][e]);
        }
#pragma unroll
        for (int j = 0; j < 4; ++j) {
            const unsigned short* bp = brow0 + j * bstride16 + kk;
            fb[j].q[0] = *(const v4u*)(bp + 8 * half);
            fb[j].q[1] = *(const v4u*)(bp + 16 + 8 * half);
        }
#pragma unroll
        for (int i = 0; i < 2; ++i)
#pragma unroll
            for (int j = 0; j < 4; ++j)
                acc[i][j] = __builtin_amdgcn_wmma_f32_16x16x32_bf16(
                    false, fa[i].bf, false, fb[j].bf, (short)0, acc[i][j],
                    false, false);
    }

#pragma unroll
    for (int i = 0; i < 2; ++i)
#pragma unroll
        for (int j = 0; j < 4; ++j)
#pragma unroll
            for (int v = 0; v < 8; ++v) {
                const int row = b * SEQ + tileM + i * 16 + v + 8 * half;
                const int col = h * HDIM + j * 16 + mn;
                concat[(size_t)row * EMBED + col] = f2bf(acc[i][j][v]);
            }
}

// ---------------------------------------------------------------------------
// launcher
// ---------------------------------------------------------------------------
extern "C" void kernel_launch(void* const* d_in, const int* in_sizes, int n_in,
                              void* d_out, int out_size, void* d_ws, size_t ws_size,
                              hipStream_t stream) {
    const float* X  = (const float*)d_in[0];
    const float* Wq = (const float*)d_in[1];
    const float* bq = (const float*)d_in[2];
    const float* Wk = (const float*)d_in[3];
    const float* bk = (const float*)d_in[4];
    const float* Wv = (const float*)d_in[5];
    const float* bv = (const float*)d_in[6];
    const float* Wo = (const float*)d_in[7];
    const float* bo = (const float*)d_in[8];

    float* out     = (float*)d_out;                       // [B,S,D]
    float* weights = out + (size_t)MTOK * EMBED;          // [B,H,S,S]

    char* ws = (char*)d_ws;
    unsigned short* Xb  = (unsigned short*)(ws);                       //  8 MB
    unsigned short* Wqt = (unsigned short*)(ws + ( 8ull << 20));       //  2 MB
    unsigned short* Wkt = (unsigned short*)(ws + (10ull << 20));       //  2 MB
    unsigned short* Wvt = (unsigned short*)(ws + (12ull << 20));       //  2 MB
    unsigned short* Wot = (unsigned short*)(ws + (14ull << 20));       //  2 MB
    unsigned short* Qb  = (unsigned short*)(ws + (16ull << 20));       //  8 MB
    unsigned short* Kb  = (unsigned short*)(ws + (24ull << 20));       //  8 MB
    unsigned short* Vt  = (unsigned short*)(ws + (32ull << 20));       //  8 MB ([EMBED, MTOK])
    unsigned short* Cc  = (unsigned short*)(ws + (40ull << 20));       //  8 MB concat

    // 1. convert activations + weights to bf16 (weights transposed)
    {
        int n = MTOK * EMBED;
        cvt_bf16_kernel<<<n / 256, 256, 0, stream>>>(X, Xb, n);
    }
    {
        dim3 tb(32, 8), tg(EMBED / 32, EMBED / 32);
        transpose_cvt_kernel<<<tg, tb, 0, stream>>>(Wq, Wqt);
        transpose_cvt_kernel<<<tg, tb, 0, stream>>>(Wk, Wkt);
        transpose_cvt_kernel<<<tg, tb, 0, stream>>>(Wv, Wvt);
        transpose_cvt_kernel<<<tg, tb, 0, stream>>>(Wo, Wot);
    }

    // 2. QKV projections (V stored transposed for the PV einsum)
    dim3 gg(EMBED / 128, MTOK / 128);
    gemm_wmma_kernel<0><<<gg, 256, 0, stream>>>(Xb, Wqt, bq, Qb, EMBED, EMBED);
    gemm_wmma_kernel<0><<<gg, 256, 0, stream>>>(Xb, Wkt, bk, Kb, EMBED, EMBED);
    gemm_wmma_kernel<1><<<gg, 256, 0, stream>>>(Xb, Wvt, bv, Vt, EMBED, MTOK);

    // 3. scores -> d_out weights region
    {
        dim3 sg(SEQ / 128, SEQ / 128, BATCH * NHEAD);
        scores_wmma_kernel<<<sg, 256, 0, stream>>>(Qb, Kb, weights);
    }

    // 4. softmax in place
    softmax_kernel<<<BATCH * NHEAD * SEQ, 256, 0, stream>>>(weights);

    // 5. P @ V -> concat (bf16)
    {
        dim3 ag(1, SEQ / 256, BATCH * NHEAD);
        av_wmma_kernel<<<ag, 256, 0, stream>>>(weights, Vt, Cc);
    }

    // 6. output projection -> fp32 d_out
    gemm_wmma_kernel<2><<<gg, 256, 0, stream>>>(Cc, Wot, bo, out, EMBED, EMBED);
}
